// KernelNN_21062519619855
// MI455X (gfx1250) — matrix operand order repaired and checked by reading, not compiled, not run on hardware
//
#include <hip/hip_runtime.h>

// ---------------------------------------------------------------------------
// NNConv GNN for MI455X (gfx1250, wave32).
// Heavy GEMMs (edge-MLP k2, k3) run on v_wmma_f32_16x16x32_f16.
// Edge weight tensor w_e kept in f16 to halve the dominant HBM stream;
// the per-depth w_e stream uses b64-vectorized loads + cross-lane reduce.
// ---------------------------------------------------------------------------

typedef __attribute__((ext_vector_type(16))) _Float16 v16h;
typedef __attribute__((ext_vector_type(8)))  _Float16 v8h;
typedef __attribute__((ext_vector_type(8)))  float    v8f;

#define NNODE 10000
#define NEDGE 120000
#define DEPTH 4

union AFrag { v16h v; v8h h[2]; };

// ---------------------------------------------------------------------------
// Generic 16x16-tile f16 GEMM:  C[M,Ntot] = act(A[M,Ktot] * BT^T + bias)
// BT is the weight matrix stored transposed: BT[n][k], f16.
// One wave per 16x16 output tile; K looped in steps of 32 (one WMMA each).
// Grid must be exact (M%16==0, waves = (M/16)*(Ntot/16), 4 waves/block).
// ---------------------------------------------------------------------------
__global__ __launch_bounds__(128) void gemm_f16_wmma(
    const _Float16* __restrict__ A,
    const _Float16* __restrict__ BT,
    const float*    __restrict__ bias,
    _Float16*       __restrict__ C,
    int Ntot, int Ktot, int doRelu)
{
    const int wave   = blockIdx.x * 4 + (threadIdx.x >> 5);
    const int lane   = threadIdx.x & 31;
    const int nTiles = Ntot >> 4;
    const int mt     = wave / nTiles;
    const int nt     = wave - mt * nTiles;
    const int m0     = mt << 4;
    const int n0     = nt << 4;
    const int hi     = lane >> 4;   // 0 or 1 (half-wave)
    const int r      = lane & 15;

    const _Float16* arow = A  + (size_t)(m0 + r) * Ktot;
    const _Float16* brow = BT + (size_t)(n0 + r) * Ktot;

    v8f c = {};
    for (int kk = 0; kk < Ktot; kk += 32) {
        // A fragment (16x32): lane<16 -> K {kk..kk+7, kk+16..kk+23}, lane>=16 shifted by 8
        AFrag a;
        a.h[0] = *(const v8h*)(arow + kk      + hi * 8);
        a.h[1] = *(const v8h*)(arow + kk + 16 + hi * 8);
        // B fragment (32x16): lane column n0+r, K {kk..kk+7, kk+16..kk+23}, lane>=16 shifted by 8
        AFrag bf;
        bf.h[0] = *(const v8h*)(brow + kk      + hi * 8);
        bf.h[1] = *(const v8h*)(brow + kk + 16 + hi * 8);
        v16h b = bf.v;
        c = __builtin_amdgcn_wmma_f32_16x16x32_f16(false, a.v, false, b,
                                                   (short)0, c, false, false);
    }

    const float bo = bias[n0 + r];
    _Float16* crow = C + (size_t)m0 * Ntot + n0 + r;
#pragma unroll
    for (int i = 0; i < 8; ++i) {
        float v = c[i] + bo;
        if (doRelu) v = v > 0.f ? v : 0.f;
        crow[(size_t)(i + hi * 8) * Ntot] = (_Float16)v;
    }
}

// W[K][Nn] f32 (row-major)  ->  WT[Nn][K] f16
__global__ void transpose_to_f16(const float* __restrict__ W,
                                 _Float16* __restrict__ WT, int K, int Nn)
{
    int idx = blockIdx.x * blockDim.x + threadIdx.x;
    if (idx >= K * Nn) return;
    int n = idx / K, k = idx - n * K;
    WT[idx] = (_Float16)W[(size_t)k * Nn + n];
}

// h1 = relu(edge_attr[E,6] @ k1_w[6,128] + b)  -> f16 [E,128]
__global__ void edge_mlp1(const float* __restrict__ ea,
                          const float* __restrict__ w,
                          const float* __restrict__ b,
                          _Float16* __restrict__ h1, int E)
{
    int idx = blockIdx.x * blockDim.x + threadIdx.x;
    int e = idx >> 7, j = idx & 127;
    if (e >= E) return;
    const float* row = ea + (size_t)e * 6;
    float acc = b[j];
#pragma unroll
    for (int i = 0; i < 6; ++i) acc += row[i] * w[i * 128 + j];
    h1[(size_t)e * 128 + j] = (_Float16)(acc > 0.f ? acc : 0.f);
}

// h = x[N,6] @ fc1_w[6,32] + b   (f32 state)
__global__ void node_fc1(const float* __restrict__ x,
                         const float* __restrict__ w,
                         const float* __restrict__ b,
                         float* __restrict__ h, int N)
{
    int idx = blockIdx.x * blockDim.x + threadIdx.x;
    int v = idx >> 5, o = idx & 31;
    if (v >= N) return;
    const float* row = x + (size_t)v * 6;
    float acc = b[o];
#pragma unroll
    for (int i = 0; i < 6; ++i) acc += row[i] * w[i * 32 + o];
    h[idx] = acc;
}

__global__ void indeg_count(const int* __restrict__ dst, int* __restrict__ deg, int E)
{
    int e = blockIdx.x * blockDim.x + threadIdx.x;
    if (e < E) atomicAdd(&deg[dst[e]], 1);
}

// ---------------------------------------------------------------------------
// Per-edge matvec msg = h[src] @ w_e[e] (32x32 f16), scatter-add into aggr.
// One wave per edge. w_e tile viewed as 256 x uint2 (b64 loads, 256 B
// contiguous per wave iteration). uint2 q covers halves 4q..4q+3 of the
// row-major tile: with q = t*32+lane -> row i = t*4 + lane/8, channels
// 4*(lane%8)..+3. Lanes {l,l+8,l+16,l+24} partition rows for one channel
// quad; reduce via shfl_xor(8),(16); lanes 0..7 scatter 4 channels each.
// ---------------------------------------------------------------------------
__global__ __launch_bounds__(256) void msg_scatter(
    const float* __restrict__ h,
    const _Float16* __restrict__ w_e,
    const int* __restrict__ src, const int* __restrict__ dst,
    float* __restrict__ aggr, int E)
{
    const int e = blockIdx.x * 8 + (threadIdx.x >> 5);
    const int lane = threadIdx.x & 31;
    if (e >= E) return;

    const float hv = h[(size_t)src[e] * 32 + lane];   // lane l holds h[src][l]
    const uint2* W2 = (const uint2*)(w_e + (size_t)e * 1024);
    const int rsel = lane >> 3;                       // row sub-group 0..3

    float a0 = 0.f, a1 = 0.f, a2 = 0.f, a3 = 0.f;
#pragma unroll
    for (int t = 0; t < 8; ++t) {
        union { uint2 u; _Float16 q[4]; } w;
        w.u = W2[t * 32 + lane];
        const float m = __shfl(hv, t * 4 + rsel, 32); // h[src][row i]
        a0 += (float)w.q[0] * m;
        a1 += (float)w.q[1] * m;
        a2 += (float)w.q[2] * m;
        a3 += (float)w.q[3] * m;
    }
    a0 += __shfl_xor(a0, 8, 32);  a0 += __shfl_xor(a0, 16, 32);
    a1 += __shfl_xor(a1, 8, 32);  a1 += __shfl_xor(a1, 16, 32);
    a2 += __shfl_xor(a2, 8, 32);  a2 += __shfl_xor(a2, 16, 32);
    a3 += __shfl_xor(a3, 8, 32);  a3 += __shfl_xor(a3, 16, 32);

    if (lane < 8) {
        float* ap = aggr + (size_t)dst[e] * 32 + lane * 4;
        atomicAdd(ap + 0, a0);
        atomicAdd(ap + 1, a1);
        atomicAdd(ap + 2, a2);
        atomicAdd(ap + 3, a3);
    }
}

// h_new = aggr/deg + h_old @ root_w + conv_b  (+relu except last depth)
__global__ void node_update(const float* __restrict__ aggr,
                            const int* __restrict__ deg,
                            const float* __restrict__ hold,
                            const float* __restrict__ rootw,
                            const float* __restrict__ cb,
                            float* __restrict__ hnew,
                            int N, int doRelu)
{
    int idx = blockIdx.x * blockDim.x + threadIdx.x;
    int v = idx >> 5, o = idx & 31;
    if (v >= N) return;
    int d = deg[v]; if (d < 1) d = 1;
    float acc = cb[o] + aggr[idx] / (float)d;
    const float* hr = hold + (size_t)v * 32;
#pragma unroll
    for (int i = 0; i < 32; ++i) acc += hr[i] * rootw[i * 32 + o];
    if (doRelu) acc = acc > 0.f ? acc : 0.f;
    hnew[idx] = acc;
}

__global__ void node_fc2(const float* __restrict__ h,
                         const float* __restrict__ w,
                         const float* __restrict__ b,
                         float* __restrict__ t, int N)
{
    int idx = blockIdx.x * blockDim.x + threadIdx.x;
    int v = idx >> 7, j = idx & 127;
    if (v >= N) return;
    const float* hr = h + (size_t)v * 32;
    float acc = b[j];
#pragma unroll
    for (int i = 0; i < 32; ++i) acc += hr[i] * w[i * 128 + j];
    t[idx] = acc > 0.f ? acc : 0.f;
}

__global__ void node_fc3(const float* __restrict__ t,
                         const float* __restrict__ w,
                         const float* __restrict__ b,
                         float* __restrict__ out, int N)
{
    int v = blockIdx.x * blockDim.x + threadIdx.x;
    if (v >= N) return;
    const float* tr = t + (size_t)v * 128;
    float acc = b[0];
#pragma unroll 16
    for (int j = 0; j < 128; ++j) acc += tr[j] * w[j];
    out[v] = acc;
}

// ---------------------------------------------------------------------------
extern "C" void kernel_launch(void* const* d_in, const int* in_sizes, int n_in,
                              void* d_out, int out_size, void* d_ws, size_t ws_size,
                              hipStream_t stream)
{
    (void)in_sizes; (void)n_in; (void)out_size; (void)ws_size;
    const float* x      = (const float*)d_in[0];
    const int*   ei     = (const int*)  d_in[1];
    const float* ea     = (const float*)d_in[2];
    const float* fc1_w  = (const float*)d_in[3];
    const float* fc1_b  = (const float*)d_in[4];
    const float* k1_w   = (const float*)d_in[5];
    const float* k1_b   = (const float*)d_in[6];
    const float* k2_w   = (const float*)d_in[7];
    const float* k2_b   = (const float*)d_in[8];
    const float* k3_w   = (const float*)d_in[9];
    const float* k3_b   = (const float*)d_in[10];
    const float* root_w = (const float*)d_in[11];
    const float* conv_b = (const float*)d_in[12];
    const float* fc2_w  = (const float*)d_in[13];
    const float* fc2_b  = (const float*)d_in[14];
    const float* fc3_w  = (const float*)d_in[15];
    const float* fc3_b  = (const float*)d_in[16];
    const int* src = ei;
    const int* dst = ei + NEDGE;
    float* out = (float*)d_out;

    // workspace carve-up (256B aligned)
    char* ws = (char*)d_ws;
    size_t off = 0;
    auto carve = [&](size_t bytes) -> char* {
        char* p = ws + off;
        off = (off + bytes + 255) & ~(size_t)255;
        return p;
    };
    _Float16* w_e  = (_Float16*)carve((size_t)NEDGE * 1024 * 2); // 246 MB
    _Float16* h1   = (_Float16*)carve((size_t)NEDGE * 128 * 2);
    _Float16* h2   = (_Float16*)carve((size_t)NEDGE * 256 * 2);
    _Float16* k2T  = (_Float16*)carve((size_t)256 * 128 * 2);
    _Float16* k3T  = (_Float16*)carve((size_t)1024 * 256 * 2);
    float*    hA   = (float*)   carve((size_t)NNODE * 32 * 4);
    float*    hB   = (float*)   carve((size_t)NNODE * 32 * 4);
    float*    aggr = (float*)   carve((size_t)NNODE * 32 * 4);
    int*      deg  = (int*)     carve((size_t)NNODE * 4);
    float*    tbuf = (float*)   carve((size_t)NNODE * 128 * 4);

    // in-degree (scatter-mean denominator)
    hipMemsetAsync(deg, 0, (size_t)NNODE * 4, stream);
    indeg_count<<<(NEDGE + 255) / 256, 256, 0, stream>>>(dst, deg, NEDGE);

    // pack weights (transposed, f16) for WMMA B operands
    transpose_to_f16<<<(256 * 128 + 255) / 256, 256, 0, stream>>>(k2_w, k2T, 128, 256);
    transpose_to_f16<<<(1024 * 256 + 255) / 256, 256, 0, stream>>>(k3_w, k3T, 256, 1024);

    // edge MLP: k1 (VALU, K=6) then k2/k3 on WMMA
    edge_mlp1<<<(NEDGE * 128) / 256, 256, 0, stream>>>(ea, k1_w, k1_b, h1, NEDGE);
    // h2 = relu(h1 @ k2): (E/16)*(256/16) waves, 4 waves/block
    gemm_f16_wmma<<<(NEDGE / 16) * (256 / 16) / 4, 128, 0, stream>>>(
        h1, k2T, k2_b, h2, 256, 128, 1);
    // w_e = h2 @ k3: (E/16)*(1024/16) waves
    gemm_f16_wmma<<<(NEDGE / 16) * (1024 / 16) / 4, 128, 0, stream>>>(
        h2, k3T, k3_b, w_e, 1024, 256, 0);

    // node init: h = x @ fc1_w + b
    node_fc1<<<(NNODE * 32 + 255) / 256, 256, 0, stream>>>(x, fc1_w, fc1_b, hA, NNODE);

    float* hc = hA;
    float* hn = hB;
    for (int d = 0; d < DEPTH; ++d) {
        hipMemsetAsync(aggr, 0, (size_t)NNODE * 32 * 4, stream);
        msg_scatter<<<(NEDGE + 7) / 8, 256, 0, stream>>>(hc, w_e, src, dst, aggr, NEDGE);
        node_update<<<(NNODE * 32 + 255) / 256, 256, 0, stream>>>(
            aggr, deg, hc, root_w, conv_b, hn, NNODE, d != DEPTH - 1);
        float* tf = hc; hc = hn; hn = tf;
    }

    node_fc2<<<(NNODE * 128 + 255) / 256, 256, 0, stream>>>(hc, fc2_w, fc2_b, tbuf, NNODE);
    node_fc3<<<(NNODE + 255) / 256, 256, 0, stream>>>(tbuf, fc3_w, fc3_b, out, NNODE);
}
